// LEDNet_5712306504529
// MI455X (gfx1250) — matrix operand order repaired and checked by reading, not compiled, not run on hardware
//
#include <hip/hip_runtime.h>
#include <hip/hip_bf16.h>

// ---------------------------------------------------------------------------
// LEDNet forward for MI455X (gfx1250, wave32, WMMA).
// Internal activations are NHWC so the implicit-GEMM A-stage is 4 aligned
// float4 loads per thread per K-chunk (one tap, one bounds check) instead of
// 16 predicated scalar gathers. v_wmma_f32_16x16x32_f16, f32 accumulate,
// bias/residual/activation fused, LDS double-buffered (1 barrier/chunk).
// ---------------------------------------------------------------------------

typedef _Float16 v16h __attribute__((ext_vector_type(16)));
typedef float    v8f  __attribute__((ext_vector_type(8)));

static __device__ __forceinline__ float activate(float v, int act, float alpha) {
    switch (act) {
        case 1: return v >= 0.f ? v : alpha * v;        // PReLU
        case 2: return v > 0.f ? v : 0.f;               // ReLU
        case 3: return 1.f / (1.f + __expf(-v));        // sigmoid
        case 4: return v >= 0.f ? v : 0.2f * v;         // leaky 0.2
    }
    return v;
}

#define CT_M 128
#define CT_N 32
#define CT_K 32

// ---------------------------------------------------------------------------
// Fast path: NHWC in, NHWC (or NCHW) out. Requires Cin multiple of 32, pow2.
//   K order: kk = tap*Cin + cin  (tap = kh*KS+kw).
// Block 256 = 8 wave32; tile 128(M) x 32(N); 2 WMMA per wave per chunk.
// ---------------------------------------------------------------------------
template <int KS>
__global__ __launch_bounds__(256)
void conv2d_wmma_nhwc(const float* __restrict__ x, const float* __restrict__ w,
                      const float* __restrict__ bias, const float* __restrict__ resid,
                      const float* __restrict__ alphaPtr, float* __restrict__ y,
                      int Cin, int H, int W, int Cout, int act, int nchwOut)
{
    constexpr int KS2 = KS * KS;
    __shared__ _Float16 As[2][CT_M * CT_K];   // [m][k]
    __shared__ _Float16 Bs[2][CT_N * CT_K];   // [n][k]

    const int tid  = threadIdx.x;
    const int lane = tid & 31;
    const int wave = tid >> 5;
    const int grp  = (lane >> 4) & 1;
    const int mr   = lane & 15;

    const int M   = H * W;
    const int pad = (KS - 1) >> 1;
    const int KK  = Cin * KS2;                 // multiple of 32
    const int lc  = __builtin_ctz(Cin);
    const int m0  = blockIdx.x * CT_M;
    const int n0  = blockIdx.y * CT_N;

    // A staging: idx = tid*16+i -> one row aRow, 16 consecutive kk
    const int aRow   = (tid * 16) >> 5;        // 0..127
    const int kiBase = (tid * 16) & 31;        // 0 or 16
    int ihb, iwb;
    bool pvalid;
    {
        int p  = m0 + aRow;
        pvalid = p < M;
        int pc = pvalid ? p : 0;
        int oh = pc / W, ow = pc - oh * W;
        ihb = oh - pad;
        iwb = ow - pad;
    }
    // B staging: idx = tid*4 -> one n, 4 consecutive kk (same tap)
    const int bN  = (tid * 4) >> 5;            // 0..31
    const int bKi = (tid * 4) & 31;            // multiple of 4

    auto stage = [&](int kb, int buf) {
        // ---- A: one tap, one bounds check, 4 float4 loads ----
        int kk0  = kb + kiBase;
        int tap  = kk0 >> lc;                  // Cin pow2
        int cin0 = kk0 & (Cin - 1);            // multiple of 16
        int kh   = tap / KS;                   // constant divisor
        int kw   = tap - kh * KS;
        int ih   = ihb + kh, iw = iwb + kw;
        bool ok  = pvalid && (unsigned)ih < (unsigned)H && (unsigned)iw < (unsigned)W;
        int base = ok ? ((ih * W + iw) << lc) + cin0 : 0;   // 64B aligned
        float4 q[4];
        #pragma unroll
        for (int j = 0; j < 4; ++j)
            q[j] = *(const float4*)&x[base + 4 * j];
        const float* qf = (const float*)q;
        #pragma unroll
        for (int i = 0; i < 16; i += 2) {
            union { unsigned u; _Float16 h[2]; } pk;
            pk.h[0] = (_Float16)(ok ? qf[i]     : 0.f);
            pk.h[1] = (_Float16)(ok ? qf[i + 1] : 0.f);
            *(unsigned*)&As[buf][aRow * CT_K + kiBase + i] = pk.u;
        }
        // ---- B: weights [Cout][Cin][KS][KS]; kk=tap*Cin+cin -> reindex ----
        int kkb  = kb + bKi;
        int tapb = kkb >> lc;
        int cb0  = kkb & (Cin - 1);
        int n    = n0 + bN;
        bool okn = n < Cout;
        int wbase = okn ? n * KK + cb0 * KS2 + tapb : 0;
        float vb[4];
        #pragma unroll
        for (int i = 0; i < 4; ++i) {
            float v = w[wbase + i * KS2];
            vb[i] = okn ? v : 0.f;
        }
        #pragma unroll
        for (int i = 0; i < 4; i += 2) {
            union { unsigned u; _Float16 h[2]; } pk;
            pk.h[0] = (_Float16)vb[i];
            pk.h[1] = (_Float16)vb[i + 1];
            *(unsigned*)&Bs[buf][bN * CT_K + bKi + i] = pk.u;
        }
        if (kb + CT_K < KK && (tid & 63) == 0)
            __builtin_prefetch(&w[n0 * KK + kb + CT_K], 0, 1);
    };

    v8f acc0 = {0.f, 0.f, 0.f, 0.f, 0.f, 0.f, 0.f, 0.f};
    v8f acc1 = {0.f, 0.f, 0.f, 0.f, 0.f, 0.f, 0.f, 0.f};

    const int nch = KK >> 5;                   // exact: KK % 32 == 0
    stage(0, 0);
    for (int c = 0; c < nch; ++c) {
        __syncthreads();
        if (c + 1 < nch) stage((c + 1) * CT_K, (c + 1) & 1);
        const int buf = c & 1;

        union { v16h v; unsigned u[8]; } a, b0, b1;
        const int arow = (wave * 16 + mr) * CT_K;
        #pragma unroll
        for (int j = 0; j < 8; ++j) {
            int k2 = 2 * j + (grp ? 16 : 0);
            a.u[j] = *(const unsigned*)&As[buf][arow + k2];
            int k2b = 2 * j + (grp ? 16 : 0);
            b0.u[j] = *(const unsigned*)&Bs[buf][mr * CT_K + k2b];
            b1.u[j] = *(const unsigned*)&Bs[buf][(mr + 16) * CT_K + k2b];
        }
        acc0 = __builtin_amdgcn_wmma_f32_16x16x32_f16(
                   false, a.v, false, b0.v, (short)0, acc0, false, false);
        acc1 = __builtin_amdgcn_wmma_f32_16x16x32_f16(
                   false, a.v, false, b1.v, (short)0, acc1, false, false);
    }

    const float alpha = alphaPtr ? alphaPtr[0] : 0.f;
    #pragma unroll
    for (int v = 0; v < 8; ++v) {
        int m = wave * 16 + v + (grp ? 8 : 0);
        int p = m0 + m;
        if (p >= M) continue;
        int n = n0 + mr;
        if (n < Cout) {
            float o = acc0[v];
            if (bias)  o += bias[n];
            int oi = nchwOut ? n * M + p : p * Cout + n;
            if (resid) o += resid[oi];
            y[oi] = activate(o, act, alpha);
        }
        n += 16;
        if (n < Cout) {
            float o = acc1[v];
            if (bias)  o += bias[n];
            int oi = nchwOut ? n * M + p : p * Cout + n;
            if (resid) o += resid[oi];
            y[oi] = activate(o, act, alpha);
        }
    }
}

// ---------------------------------------------------------------------------
// Stem conv (Cin=3, NCHW input from the harness) -> NHWC output. Generic
// scalar im2col gather; runs once so efficiency is secondary.
// ---------------------------------------------------------------------------
__global__ __launch_bounds__(256)
void conv2d_wmma_stem(const float* __restrict__ x, const float* __restrict__ w,
                      const float* __restrict__ bias, const float* __restrict__ alphaPtr,
                      float* __restrict__ y, int Cin, int H, int W, int Cout, int act)
{
    __shared__ _Float16 As[CT_M * CT_K];
    __shared__ _Float16 Bs[CT_N * CT_K];

    const int tid  = threadIdx.x;
    const int lane = tid & 31;
    const int wave = tid >> 5;
    const int grp  = (lane >> 4) & 1;
    const int mr   = lane & 15;

    const int M  = H * W;
    const int KK = Cin * 9;
    const int m0 = blockIdx.x * CT_M;
    const int n0 = blockIdx.y * CT_N;

    const int aRow   = (tid * 16) >> 5;
    const int kiBase = (tid * 16) & 31;
    int ihb, iwb; bool pvalid;
    {
        int p = m0 + aRow;
        pvalid = p < M;
        int pc = pvalid ? p : 0;
        int oh = pc / W, ow = pc - oh * W;
        ihb = oh - 1; iwb = ow - 1;
    }
    const int bN  = (tid * 4) >> 5;
    const int bKi = (tid * 4) & 31;

    v8f acc0 = {0.f,0.f,0.f,0.f,0.f,0.f,0.f,0.f};
    v8f acc1 = {0.f,0.f,0.f,0.f,0.f,0.f,0.f,0.f};

    for (int kb = 0; kb < KK; kb += CT_K) {
        float va[16];
        #pragma unroll
        for (int i = 0; i < 16; ++i) {
            int kk  = kb + kiBase + i;
            int cin = kk / 9;
            int rem = kk - cin * 9;
            int kh  = rem / 3;
            int kw  = rem - kh * 3;
            int ih  = ihb + kh, iw = iwb + kw;
            bool ok = pvalid && kk < KK &&
                      (unsigned)ih < (unsigned)H && (unsigned)iw < (unsigned)W;
            int off = (cin * H + ih) * W + iw;
            float v = x[ok ? off : 0];
            va[i] = ok ? v : 0.f;
        }
        #pragma unroll
        for (int i = 0; i < 16; i += 2) {
            union { unsigned u; _Float16 h[2]; } pk;
            pk.h[0] = (_Float16)va[i];
            pk.h[1] = (_Float16)va[i + 1];
            *(unsigned*)&As[aRow * CT_K + kiBase + i] = pk.u;
        }
        {
            int n = n0 + bN;
            int kk = kb + bKi;
            #pragma unroll
            for (int i = 0; i < 4; i += 2) {
                union { unsigned u; _Float16 h[2]; } pk;
                bool o0 = n < Cout && kk + i < KK;
                bool o1 = n < Cout && kk + i + 1 < KK;
                float v0 = w[o0 ? n * KK + kk + i : 0];
                float v1 = w[o1 ? n * KK + kk + i + 1 : 0];
                pk.h[0] = (_Float16)(o0 ? v0 : 0.f);
                pk.h[1] = (_Float16)(o1 ? v1 : 0.f);
                *(unsigned*)&Bs[bN * CT_K + bKi + i] = pk.u;
            }
        }
        __syncthreads();
        union { v16h v; unsigned u[8]; } a, b0, b1;
        const int arow = (wave * 16 + mr) * CT_K;
        #pragma unroll
        for (int j = 0; j < 8; ++j) {
            int k2 = 2 * j + (grp ? 16 : 0);
            a.u[j] = *(const unsigned*)&As[arow + k2];
            int k2b = 2 * j + (grp ? 16 : 0);
            b0.u[j] = *(const unsigned*)&Bs[mr * CT_K + k2b];
            b1.u[j] = *(const unsigned*)&Bs[(mr + 16) * CT_K + k2b];
        }
        acc0 = __builtin_amdgcn_wmma_f32_16x16x32_f16(
                   false, a.v, false, b0.v, (short)0, acc0, false, false);
        acc1 = __builtin_amdgcn_wmma_f32_16x16x32_f16(
                   false, a.v, false, b1.v, (short)0, acc1, false, false);
        __syncthreads();
    }
    const float alpha = alphaPtr ? alphaPtr[0] : 0.f;
    #pragma unroll
    for (int v = 0; v < 8; ++v) {
        int m = wave * 16 + v + (grp ? 8 : 0);
        int p = m0 + m;
        if (p >= M) continue;
        int n = n0 + mr;
        if (n < Cout) {
            float o = acc0[v] + (bias ? bias[n] : 0.f);
            y[p * Cout + n] = activate(o, act, alpha);
        }
        n += 16;
        if (n < Cout) {
            float o = acc1[v] + (bias ? bias[n] : 0.f);
            y[p * Cout + n] = activate(o, act, alpha);
        }
    }
}

// ---------------------------------------------------------------------------
// NHWC elementwise / sampling kernels (channel index fastest everywhere).
// ---------------------------------------------------------------------------
__global__ void blur_down_k(const float* __restrict__ x, float* __restrict__ y,
                            int C, int H, int W)
{
    int OH = H >> 1, OW = W >> 1;
    int idx = blockIdx.x * blockDim.x + threadIdx.x;
    int total = C * OH * OW;
    if (idx >= total) return;
    int c = idx % C; int p2 = idx / C; int ow = p2 % OW; int oh = p2 / OW;
    const float kf[3] = {0.25f, 0.5f, 0.25f};
    float s = 0.f;
    for (int a = 0; a < 3; ++a) {
        int ih = 2 * oh + a - 1;
        ih = ih < 0 ? -ih : (ih >= H ? 2 * H - 2 - ih : ih);
        float rs = 0.f;
        for (int b = 0; b < 3; ++b) {
            int iw = 2 * ow + b - 1;
            iw = iw < 0 ? -iw : (iw >= W ? 2 * W - 2 - iw : iw);
            rs += kf[b] * x[(ih * W + iw) * C + c];
        }
        s += kf[a] * rs;
    }
    y[idx] = s;
}

__global__ void convt3s2_prelu_k(const float* __restrict__ x,
                                 const float* __restrict__ w,
                                 const float* __restrict__ alphaPtr,
                                 float* __restrict__ y, int C, int H, int W)
{
    int OH = 2 * H, OW = 2 * W;
    int idx = blockIdx.x * blockDim.x + threadIdx.x;
    int total = C * OH * OW;
    if (idx >= total) return;
    int co = idx % C; int p2 = idx / C; int ow = p2 % OW; int oh = p2 / OW;
    float alpha = alphaPtr[0];
    float s = 0.f;
    for (int kh = 0; kh < 3; ++kh) {
        int jh = oh + 1 - kh;
        if ((jh & 1) || jh < 0) continue;
        int ih = jh >> 1; if (ih >= H) continue;
        for (int kw = 0; kw < 3; ++kw) {
            int jw = ow + 1 - kw;
            if ((jw & 1) || jw < 0) continue;
            int iw = jw >> 1; if (iw >= W) continue;
            const float* xp = x + (ih * W + iw) * C;       // contiguous in ci
            const float* wp = w + co * 9 + kh * 3 + kw;
            for (int ci = 0; ci < C; ++ci)
                s += wp[ci * C * 9] * xp[ci];
        }
    }
    y[idx] = s >= 0.f ? s : alpha * s;
}

__global__ void resize_k(const float* __restrict__ x, float* __restrict__ y,
                         int C, int H, int W, int OH, int OW, int align,
                         int Ctot, int Coff)
{
    int idx = blockIdx.x * blockDim.x + threadIdx.x;
    int total = C * OH * OW;
    if (idx >= total) return;
    int c = idx % C; int p2 = idx / C; int ox = p2 % OW; int oy = p2 / OW;
    float fy, fx;
    if (align) {
        fy = (OH > 1) ? oy * (float)(H - 1) / (float)(OH - 1) : 0.f;
        fx = (OW > 1) ? ox * (float)(W - 1) / (float)(OW - 1) : 0.f;
    } else {
        fy = fminf(fmaxf((oy + 0.5f) * (float)H / (float)OH - 0.5f, 0.f), (float)(H - 1));
        fx = fminf(fmaxf((ox + 0.5f) * (float)W / (float)OW - 0.5f, 0.f), (float)(W - 1));
    }
    int y0 = (int)floorf(fy); int y1 = y0 + 1 < H ? y0 + 1 : H - 1; float wy = fy - y0;
    int x0 = (int)floorf(fx); int x1 = x0 + 1 < W ? x0 + 1 : W - 1; float wx = fx - x0;
    float a = x[(y0 * W + x0) * C + c] * (1.f - wy) + x[(y1 * W + x0) * C + c] * wy;
    float b = x[(y0 * W + x1) * C + c] * (1.f - wy) + x[(y1 * W + x1) * C + c] * wy;
    y[(oy * OW + ox) * Ctot + Coff + c] = a * (1.f - wx) + b * wx;
}

__global__ void concat_copy_k(const float* __restrict__ x, float* __restrict__ y,
                              int C, int HW, int Ctot)
{
    int idx = blockIdx.x * blockDim.x + threadIdx.x;
    if (idx >= C * HW) return;
    int c = idx % C; int p = idx / C;
    y[p * Ctot + c] = x[idx];
}

__global__ void avgpool_k(const float* __restrict__ x, float* __restrict__ y,
                          int C, int H, int W, int b)
{
    int idx = blockIdx.x * blockDim.x + threadIdx.x;
    int total = C * b * b;
    if (idx >= total) return;
    int c = idx % C; int p2 = idx / C; int bx = p2 % b; int by = p2 / b;
    int hh = H / b, ww = W / b;
    float s = 0.f;
    for (int i = 0; i < hh; ++i)
        for (int j = 0; j < ww; ++j)
            s += x[((by * hh + i) * W + bx * ww + j) * C + c];
    y[idx] = s / (float)(hh * ww);
}

// Fused kernel-prediction (1x1 conv C -> 25*C) + KernelConv2D + leaky(0.2).
__global__ void fac_kc_k(const float* __restrict__ feat, const float* __restrict__ hid,
                         const float* __restrict__ w3, const float* __restrict__ b3,
                         float* __restrict__ y, int C, int H, int W)
{
    int idx = blockIdx.x * blockDim.x + threadIdx.x;
    int total = C * H * W;
    if (idx >= total) return;
    int c = idx % C; int pix = idx / C; int xw = pix % W; int yh = pix / W;

    float f[25];
    float acc = 0.f;
    #pragma unroll
    for (int tap = 0; tap < 25; ++tap) {
        int di = tap / 5 - 2, dj = tap % 5 - 2;
        int ih = yh + di; ih = ih < 0 ? 0 : (ih >= H ? H - 1 : ih);
        int iw = xw + dj; iw = iw < 0 ? 0 : (iw >= W ? W - 1 : iw);
        f[tap] = feat[(ih * W + iw) * C + c];
        acc += f[tap] * b3[c * 25 + tap];
    }
    const float* hp = hid + pix * C;           // contiguous in ci
    for (int ci = 0; ci < C; ++ci) {
        float h = hp[ci];
        const float* wr = w3 + c * 25 * C + ci;
        float s = 0.f;
        #pragma unroll
        for (int tap = 0; tap < 25; ++tap)
            s += f[tap] * wr[tap * C];
        acc += h * s;
    }
    y[idx] = acc >= 0.f ? acc : 0.2f * acc;
}

// Curve tail: clamp to [0,1], 3 iterations x += a_i * x * (1-x). a: [HW,3].
__global__ void curve_apply_k(const float* __restrict__ x, const float* __restrict__ a,
                              float* __restrict__ y, int C, int HW)
{
    int idx = blockIdx.x * blockDim.x + threadIdx.x;
    if (idx >= C * HW) return;
    int pix = idx / C;
    float v = fminf(fmaxf(x[idx], 0.f), 1.f);
    #pragma unroll
    for (int i = 0; i < 3; ++i) {
        float ai = a[pix * 3 + i];
        v = v + ai * v * (1.f - v);
    }
    y[idx] = v;
}

// ===========================================================================
// Host orchestration (all internal tensors NHWC)
// ===========================================================================
static inline int cdiv(int a, int b) { return (a + b - 1) / b; }

struct BEp { const float *b1c1_w,*b1c1_b,*b1_a,*b1c2_w,*b1c2_b,*b2c1_w,*b2c1_b,*b2_a;
             const float *t1_w,*t1_a,*t2_w,*t2_a,*t3_w,*bot_w; };
struct BDp { const float *t1_w,*t1_a,*tct_w,*t2_a,*t3_w,*bot_w;
             const float *b1c1_w,*b1c1_b,*b1_a,*b1c2_w,*b1c2_b,
                         *b2c1_w,*b2c1_b,*b2_a,*b2c2_w,*b2c2_b; };
struct PPMp { const float *cw[4],*ca[4],*fuse_w,*fuse_a; };
struct CURp { const float *c1_w,*c1_b,*c2_w,*c2_b,*c3_w,*c3_b; };
struct FACp { const float *cw[3],*cb[3],*c3_w,*c3_b; };

extern "C" void kernel_launch(void* const* d_in, const int* in_sizes, int n_in,
                              void* d_out, int out_size, void* d_ws, size_t ws_size,
                              hipStream_t stream)
{
    (void)in_sizes; (void)out_size; (void)ws_size;
    const float* x = (const float*)d_in[0];

    int ip = 1;
    auto P = [&]() -> const float* {
        const float* p = (const float*)d_in[ip < n_in ? ip : n_in - 1];
        ++ip; return p;
    };
    auto loadBE = [&](BEp& e, bool down) {
        e.b1c1_w=P(); e.b1c1_b=P(); e.b1_a=P(); e.b1c2_w=P(); e.b1c2_b=P();
        e.b2c1_w=P(); e.b2c1_b=P(); e.b2_a=P();
        if (down) { e.t1_w=P(); e.t1_a=P(); e.t2_w=P(); e.t2_a=P(); e.t3_w=P(); e.bot_w=P(); }
        else { e.t1_w=e.t1_a=e.t2_w=e.t2_a=e.t3_w=e.bot_w=nullptr; }
    };
    auto loadBD = [&](BDp& d, bool up) {
        if (up) { d.t1_w=P(); d.t1_a=P(); d.tct_w=P(); d.t2_a=P(); d.t3_w=P(); d.bot_w=P(); }
        else { d.t1_w=d.t1_a=d.tct_w=d.t2_a=d.t3_w=d.bot_w=nullptr; }
        d.b1c1_w=P(); d.b1c1_b=P(); d.b1_a=P(); d.b1c2_w=P(); d.b1c2_b=P();
        d.b2c1_w=P(); d.b2c1_b=P(); d.b2_a=P(); d.b2c2_w=P(); d.b2c2_b=P();
    };
    auto loadPPM = [&](PPMp& p) {
        for (int i = 0; i < 4; ++i) { p.cw[i]=P(); p.ca[i]=P(); }
        p.fuse_w=P(); p.fuse_a=P();
    };
    auto loadCUR = [&](CURp& c) {
        c.c1_w=P(); c.c1_b=P(); c.c2_w=P(); c.c2_b=P(); c.c3_w=P(); c.c3_b=P();
    };
    auto loadFAC = [&](FACp& f) {
        for (int i = 0; i < 3; ++i) { f.cw[i]=P(); f.cb[i]=P(); }
        f.c3_w=P(); f.c3_b=P();
    };

    const float *e1_c_w = P(), *e1_c_b = P(), *e1_a = P();
    BEp e1b; loadBE(e1b, true);
    BEp e2b; loadBE(e2b, true);
    BEp e3b; loadBE(e3b, true);
    PPMp pp1; loadPPM(pp1);
    PPMp pp2; loadPPM(pp2);
    PPMp pp3; loadPPM(pp3);
    CURp cu1; loadCUR(cu1);
    CURp cu2; loadCUR(cu2);
    CURp cu3; loadCUR(cu3);
    BEp m1p; loadBE(m1p, false);
    BEp m2p; loadBE(m2p, false);
    FACp f3p; loadFAC(f3p);
    FACp f2p; loadFAC(f2p);
    FACp f1p; loadFAC(f1p);
    BDp d3p; loadBD(d3p, false);
    BDp d2p; loadBD(d2p, true);
    BDp d1p; loadBD(d1p, true);
    BDp d0p; loadBD(d0p, true);
    const float *out_w = P(), *out_b = P();

    size_t off = 0;
    auto alloc = [&](size_t nfl) -> float* {
        float* p = (float*)((char*)d_ws + off);
        off += ((nfl * sizeof(float)) + 255) & ~(size_t)255;
        return p;
    };
    const size_t BIG = (size_t)64 * 384 * 384;
    float* T[6]; for (int i = 0; i < 6; ++i) T[i] = alloc(BIG);
    float* e1s = alloc((size_t)64  * 192 * 192);
    float* e2s = alloc((size_t)128 *  96 *  96);
    float* e3s = alloc((size_t)128 *  48 *  48);
    float* pool0 = alloc(128 * 36);
    float* pool1 = alloc(128 * 36);

    auto conv = [&](const float* xi, const float* w, const float* bias,
                    const float* resid, const float* alpha, float* y,
                    int Cin, int H, int W, int Cout, int Ks, int act,
                    int nchwOut = 0) {
        dim3 g(cdiv(H * W, CT_M), cdiv(Cout, CT_N));
        switch (Ks) {
            case 1: conv2d_wmma_nhwc<1><<<g, 256, 0, stream>>>(
                        xi, w, bias, resid, alpha, y, Cin, H, W, Cout, act, nchwOut); break;
            case 3: conv2d_wmma_nhwc<3><<<g, 256, 0, stream>>>(
                        xi, w, bias, resid, alpha, y, Cin, H, W, Cout, act, nchwOut); break;
            default: conv2d_wmma_nhwc<5><<<g, 256, 0, stream>>>(
                        xi, w, bias, resid, alpha, y, Cin, H, W, Cout, act, nchwOut); break;
        }
    };
    auto blur = [&](const float* xi, float* y, int C, int H, int W) {
        int tot = C * (H / 2) * (W / 2);
        blur_down_k<<<cdiv(tot, 256), 256, 0, stream>>>(xi, y, C, H, W);
    };
    auto convT = [&](const float* xi, const float* w, const float* alpha,
                     float* y, int C, int H, int W) {
        int tot = C * 4 * H * W;
        convt3s2_prelu_k<<<cdiv(tot, 256), 256, 0, stream>>>(xi, w, alpha, y, C, H, W);
    };
    auto resize = [&](const float* xi, float* y, int C, int H, int W,
                      int OH, int OW, int align, int Ctot, int Coff) {
        int tot = C * OH * OW;
        resize_k<<<cdiv(tot, 256), 256, 0, stream>>>(xi, y, C, H, W, OH, OW,
                                                     align, Ctot, Coff);
    };

    auto basic_e = [&](const float* in, float* out, const BEp& p, int C,
                       int H, int W, int Cout, bool down,
                       float* t1, float* t2, float* t3, float* t4) {
        conv(in, p.b1c1_w, p.b1c1_b, nullptr, p.b1_a, t1, C, H, W, C, 3, 1);
        conv(t1, p.b1c2_w, p.b1c2_b, in, nullptr, t2, C, H, W, C, 3, 0);
        conv(t2, p.b2c1_w, p.b2c1_b, nullptr, p.b2_a, down ? t1 : out, C, H, W, C, 3, 1);
        if (down) {
            conv(t1, p.t1_w, nullptr, nullptr, p.t1_a, t2, C, H, W, C, 1, 1);
            conv(t2, p.t2_w, nullptr, nullptr, p.t2_a, t3, C, H, W, C, 3, 1);
            blur(t3, t2, C, H, W);
            conv(t2, p.t3_w, nullptr, nullptr, nullptr, t4, C, H / 2, W / 2, Cout, 1, 0);
            blur(t1, t2, C, H, W);
            conv(t2, p.bot_w, nullptr, t4, nullptr, out, C, H / 2, W / 2, Cout, 1, 0);
        }
    };
    auto ppm = [&](const float* in, float* out, const PPMp& p, int C, int H, int W,
                   float* cc) {
        int r = C / 4;
        int tot = C * H * W;
        concat_copy_k<<<cdiv(tot, 256), 256, 0, stream>>>(in, cc, C, H * W, 2 * C);
        const int bins[4] = {1, 2, 3, 6};
        for (int i = 0; i < 4; ++i) {
            int b = bins[i];
            int pt = C * b * b;
            avgpool_k<<<cdiv(pt, 256), 256, 0, stream>>>(in, pool0, C, H, W, b);
            conv(pool0, p.cw[i], nullptr, nullptr, p.ca[i], pool1, C, b, b, r, 1, 1);
            resize(pool1, cc, r, b, b, H, W, 1, 2 * C, C + i * r);
        }
        conv(cc, p.fuse_w, nullptr, nullptr, p.fuse_a, out, 2 * C, H, W, C, 3, 1);
    };
    auto curve = [&](const float* in, float* out, const CURp& p, int C, int H, int W,
                     float* t1, float* t2) {
        conv(in, p.c1_w, p.c1_b, nullptr, nullptr, t1, C, H, W, C, 5, 2);
        conv(t1, p.c2_w, p.c2_b, nullptr, nullptr, t2, C, H, W, C, 3, 2);
        conv(t2, p.c3_w, p.c3_b, nullptr, nullptr, t1, C, H, W, 3, 1, 3);
        int tot = C * H * W;
        curve_apply_k<<<cdiv(tot, 256), 256, 0, stream>>>(in, t1, out, C, H * W);
    };
    auto basic_d = [&](const float* in, float* out, const BDp& p, int Cin,
                       int H, int W, int Cout, bool up,
                       float* t0, float* t1, float* t2, float* t3) {
        const float* xx = in; int C = Cout, h = H, w = W;
        if (up) {
            conv(in, p.t1_w, nullptr, nullptr, p.t1_a, t0, Cin, H, W, Cin, 1, 1);
            convT(t0, p.tct_w, p.t2_a, t1, Cin, H, W);
            h = 2 * H; w = 2 * W;
            conv(t1, p.t3_w, nullptr, nullptr, nullptr, t0, Cin, h, w, Cout, 1, 0);
            resize(in, t1, Cin, H, W, h, w, 0, Cin, 0);
            conv(t1, p.bot_w, nullptr, t0, nullptr, t2, Cin, h, w, Cout, 1, 0);
            xx = t2;
        }
        conv(xx, p.b1c1_w, p.b1c1_b, nullptr, p.b1_a, t0, C, h, w, C, 3, 1);
        conv(t0, p.b1c2_w, p.b1c2_b, xx, nullptr, t1, C, h, w, C, 3, 0);
        conv(t1, p.b2c1_w, p.b2c1_b, nullptr, p.b2_a, t0, C, h, w, C, 3, 1);
        conv(t0, p.b2c2_w, p.b2c2_b, t1, nullptr, out, C, h, w, C, 3, 0);
    };
    auto fac_kc = [&](const float* enc, const float* feat, float* out, const FACp& p,
                      int C, int H, int W, float* t0, float* t1) {
        conv(enc, p.cw[0], p.cb[0], nullptr, nullptr, t0, C, H, W, C, 3, 2);
        conv(t0,  p.cw[1], p.cb[1], nullptr, nullptr, t1, C, H, W, C, 3, 2);
        conv(t1,  p.cw[2], p.cb[2], nullptr, nullptr, t0, C, H, W, C, 3, 2);
        int tot = C * H * W;
        fac_kc_k<<<cdiv(tot, 256), 256, 0, stream>>>(feat, t0, p.c3_w, p.c3_b,
                                                     out, C, H, W);
    };

    // ---- forward (internal NHWC) ----
    {   // stem: NCHW input -> NHWC activations
        dim3 g(cdiv(384 * 384, CT_M), cdiv(32, CT_N));
        conv2d_wmma_stem<<<g, 256, 0, stream>>>(x, e1_c_w, e1_c_b, e1_a, T[0],
                                                3, 384, 384, 32, 1);
    }
    basic_e(T[0], T[5], e1b, 32, 384, 384, 64, true, T[1], T[2], T[3], T[4]);
    ppm(T[5], T[0], pp1, 64, 192, 192, T[1]);
    curve(T[0], e1s, cu1, 64, 192, 192, T[2], T[3]);

    basic_e(e1s, T[5], e2b, 64, 192, 192, 128, true, T[0], T[1], T[2], T[3]);
    ppm(T[5], T[0], pp2, 128, 96, 96, T[1]);
    curve(T[0], e2s, cu2, 128, 96, 96, T[2], T[3]);

    basic_e(e2s, T[5], e3b, 128, 96, 96, 128, true, T[0], T[1], T[2], T[3]);
    ppm(T[5], T[0], pp3, 128, 48, 48, T[1]);
    curve(T[0], e3s, cu3, 128, 48, 48, T[2], T[3]);

    basic_e(e3s, T[0], m1p, 128, 48, 48, 128, false, T[1], T[2], T[3], T[4]);
    basic_e(T[0], T[5], m2p, 128, 48, 48, 128, false, T[1], T[2], T[3], T[4]);

    basic_d(T[5], T[4], d3p, 128, 48, 48, 128, false, T[0], T[1], T[2], T[3]);
    fac_kc(e3s, T[4], T[5], f3p, 128, 48, 48, T[0], T[1]);

    basic_d(T[5], T[4], d2p, 128, 48, 48, 128, true, T[0], T[1], T[2], T[3]);
    fac_kc(e2s, T[4], T[5], f2p, 128, 96, 96, T[0], T[1]);

    basic_d(T[5], T[4], d1p, 128, 96, 96, 64, true, T[0], T[1], T[2], T[3]);
    fac_kc(e1s, T[4], T[5], f1p, 64, 192, 192, T[0], T[1]);

    basic_d(T[5], T[4], d0p, 64, 192, 192, 32, true, T[0], T[1], T[2], T[3]);

    // final conv: NHWC in, NCHW out straight into d_out
    conv(T[4], out_w, out_b, nullptr, nullptr, (float*)d_out,
         32, 384, 384, 3, 3, 0, /*nchwOut=*/1);
}
